// GSS_66709432042229
// MI455X (gfx1250) — compile-verified, matrix operand order checked by reading, not run on hardware
//
#include <hip/hip_runtime.h>
#include <hip/hip_bf16.h>
#include <hip/hip_fp16.h>

typedef __attribute__((ext_vector_type(16))) _Float16 v16h;
typedef __attribute__((ext_vector_type(8)))  float    v8f;

#define D_DIM 512

// ---------- row-sum normalize: one wave32 per row ----------
__global__ void rownorm_kernel(const float* __restrict__ x, float* __restrict__ y, int nrows) {
  int wave = (int)((blockIdx.x * blockDim.x + threadIdx.x) >> 5);
  int lane = threadIdx.x & 31;
  if (wave >= nrows) return;
  const float4* xr = (const float4*)(x + (size_t)wave * D_DIM);
  float4 v[4];
  float s = 0.f;
#pragma unroll
  for (int j = 0; j < 4; ++j) {
    v[j] = xr[lane + 32 * j];
    s += v[j].x + v[j].y + v[j].z + v[j].w;
  }
#pragma unroll
  for (int off = 16; off > 0; off >>= 1) s += __shfl_xor(s, off, 32);
  float inv = 1.0f / (s + 1e-4f);
  float4* yr = (float4*)(y + (size_t)wave * D_DIM);
#pragma unroll
  for (int j = 0; j < 4; ++j) {
    float4 o = v[j];
    o.x *= inv; o.y *= inv; o.z *= inv; o.w *= inv;
    yr[lane + 32 * j] = o;
  }
}

// ---------- L2 normalize + convert to f16: one wave32 per row ----------
__global__ void l2norm_f16_kernel(const float* __restrict__ x, _Float16* __restrict__ y, int nrows) {
  int wave = (int)((blockIdx.x * blockDim.x + threadIdx.x) >> 5);
  int lane = threadIdx.x & 31;
  if (wave >= nrows) return;
  const float4* xr = (const float4*)(x + (size_t)wave * D_DIM);
  float4 v[4];
  float s = 0.f;
#pragma unroll
  for (int j = 0; j < 4; ++j) {
    v[j] = xr[lane + 32 * j];
    s += v[j].x * v[j].x + v[j].y * v[j].y + v[j].z * v[j].z + v[j].w * v[j].w;
  }
#pragma unroll
  for (int off = 16; off > 0; off >>= 1) s += __shfl_xor(s, off, 32);
  float inv = 1.0f / fmaxf(sqrtf(s), 1e-12f);
  _Float16* yr = y + (size_t)wave * D_DIM;
#pragma unroll
  for (int j = 0; j < 4; ++j) {
    int base = (lane + 32 * j) * 4;
    yr[base + 0] = (_Float16)(v[j].x * inv);
    yr[base + 1] = (_Float16)(v[j].y * inv);
    yr[base + 2] = (_Float16)(v[j].z * inv);
    yr[base + 3] = (_Float16)(v[j].w * inv);
  }
}

// ---------- zero float4 ----------
__global__ void zero4_kernel(float4* __restrict__ p, size_t n4) {
  size_t i = (size_t)blockIdx.x * blockDim.x + threadIdx.x;
  if (i < n4) p[i] = make_float4(0.f, 0.f, 0.f, 0.f);
}

// ---------- f32 -> f16 convert ----------
__global__ void cvt_f16_kernel(const float* __restrict__ x, _Float16* __restrict__ y, size_t n) {
  size_t i = (size_t)blockIdx.x * blockDim.x + threadIdx.x;
  if (i < n) y[i] = (_Float16)x[i];
}

// ---------- SpMM scatter: 2 edges per 256-thread block, float4 per thread ----------
__global__ void spmm_kernel(const float* __restrict__ xn, const int* __restrict__ ei,
                            const float* __restrict__ ew, float* __restrict__ agg, int E) {
  int edge = blockIdx.x * 2 + (threadIdx.x >> 7);
  if (edge >= E) return;
  int t   = threadIdx.x & 127;          // 128 threads * float4 = 512 features
  int dst = ei[edge];                   // row (destination)
  int src = ei[E + edge];               // col (source)
  float w = ew[edge];
  float4 v = ((const float4*)(xn + (size_t)src * D_DIM))[t];
  float* ad = agg + (size_t)dst * D_DIM + t * 4;
  atomicAdd(ad + 0, w * v.x);
  atomicAdd(ad + 1, w * v.y);
  atomicAdd(ad + 2, w * v.z);
  atomicAdd(ad + 3, w * v.w);
}

// ---------- fragment = two b128 loads (halfs [0..7] and [16..23] of the lane's K window)
struct frag16 { uint4 lo, hi; };

__device__ __forceinline__ frag16 ldfrag(const _Float16* __restrict__ row, int kk, int hi8) {
  const uint4* p = (const uint4*)(row + kk + hi8);   // 16B aligned
  frag16 f;
  f.lo = p[0];   // halfs 0..7   (K pairs 0,2,4,6 + p)
  f.hi = p[2];   // halfs 16..23 (K pairs 16,18,20,22 + p)
  return f;
}

__device__ __forceinline__ v16h fragh(const frag16& f) {
  union { frag16 f; v16h h; } u;
  u.f = f;
  return u.h;
}

// ---------- WMMA GEMM: C[M,Nc] = act(A[M,K] * B[Nc,K]^T + bias) ----------
// K = 512. One wave computes a 32x64 C block: 2 M-tiles x 4 N-tiles,
// A fragments reused over 4 N-tiles, B fragments reused over 2 M-tiles.
// All 6 fragments of a K-step are issued before the 8 WMMAs consume them,
// letting the scheduler overlap loads with matrix math.
// ACT: 1 = bias + ELU, 2 = ReLU (no bias)
// Fragment layouts per CDNA5 ISA 7.12.2 (16-bit A 16x32, f32 C/D 16x16).
template <int ACT>
__global__ void wmma_gemm_abT_kernel(const _Float16* __restrict__ A,
                                     const _Float16* __restrict__ B,
                                     const float* __restrict__ bias,
                                     float* __restrict__ C,
                                     int M, int Nc) {
  const int K = D_DIM;
  int lane   = threadIdx.x & 31;
  int gwave  = blockIdx.x * (blockDim.x >> 5) + (threadIdx.x >> 5);
  int nt64   = Nc >> 6;
  int tm     = gwave / nt64;            // 32-row tile index
  int tn     = gwave % nt64;            // 64-col tile index
  if (tm >= (M >> 5)) return;
  int mBase = tm << 5;
  int nBase = tn << 6;
  int l15   = lane & 15;
  int hi8   = (lane >> 4) << 3;         // 0 or 8 halfs

  const _Float16* Arow0 = A + (size_t)(mBase + l15) * K;
  const _Float16* Arow1 = Arow0 + (size_t)16 * K;
  const _Float16* Brow0 = B + (size_t)(nBase + 0 * 16 + l15) * K;
  const _Float16* Brow1 = B + (size_t)(nBase + 1 * 16 + l15) * K;
  const _Float16* Brow2 = B + (size_t)(nBase + 2 * 16 + l15) * K;
  const _Float16* Brow3 = B + (size_t)(nBase + 3 * 16 + l15) * K;

  v8f acc[8];
#pragma unroll
  for (int j = 0; j < 8; ++j) acc[j] = (v8f){};

#pragma unroll 2
  for (int kk = 0; kk < K; kk += 32) {
    // issue all loads for this K-step first
    frag16 fa0 = ldfrag(Arow0, kk, hi8);
    frag16 fa1 = ldfrag(Arow1, kk, hi8);
    frag16 fb0 = ldfrag(Brow0, kk, hi8);
    frag16 fb1 = ldfrag(Brow1, kk, hi8);
    frag16 fb2 = ldfrag(Brow2, kk, hi8);
    frag16 fb3 = ldfrag(Brow3, kk, hi8);

    v16h a0 = fragh(fa0), a1 = fragh(fa1);
    v16h b0 = fragh(fb0), b1 = fragh(fb1), b2 = fragh(fb2), b3 = fragh(fb3);

    acc[0] = __builtin_amdgcn_wmma_f32_16x16x32_f16(false, a0, false, b0, (short)0, acc[0], false, false);
    acc[4] = __builtin_amdgcn_wmma_f32_16x16x32_f16(false, a1, false, b0, (short)0, acc[4], false, false);
    acc[1] = __builtin_amdgcn_wmma_f32_16x16x32_f16(false, a0, false, b1, (short)0, acc[1], false, false);
    acc[5] = __builtin_amdgcn_wmma_f32_16x16x32_f16(false, a1, false, b1, (short)0, acc[5], false, false);
    acc[2] = __builtin_amdgcn_wmma_f32_16x16x32_f16(false, a0, false, b2, (short)0, acc[2], false, false);
    acc[6] = __builtin_amdgcn_wmma_f32_16x16x32_f16(false, a1, false, b2, (short)0, acc[6], false, false);
    acc[3] = __builtin_amdgcn_wmma_f32_16x16x32_f16(false, a0, false, b3, (short)0, acc[3], false, false);
    acc[7] = __builtin_amdgcn_wmma_f32_16x16x32_f16(false, a1, false, b3, (short)0, acc[7], false, false);
  }

  // Epilogue: C/D layout — n = lane&15 (+16j), VGPR r -> row m = r + 8*(lane>=16)
  int hi = lane >> 4;
#pragma unroll
  for (int half = 0; half < 2; ++half) {       // which M-tile (rows mBase / mBase+16)
#pragma unroll
    for (int j = 0; j < 4; ++j) {
      int n = nBase + j * 16 + l15;
      float bv = (ACT == 1) ? bias[n] : 0.f;
      const v8f& a = acc[half * 4 + j];
#pragma unroll
      for (int r = 0; r < 8; ++r) {
        int m = mBase + half * 16 + r + (hi << 3);
        float z = a[r] + bv;
        if (ACT == 1) z = (z > 0.f) ? z : (expf(z) - 1.f);
        if (ACT == 2) z = fmaxf(z, 0.f);
        C[(size_t)m * Nc + n] = z;
      }
    }
  }
}

extern "C" void kernel_launch(void* const* d_in, const int* in_sizes, int n_in,
                              void* d_out, int out_size, void* d_ws, size_t ws_size,
                              hipStream_t stream) {
  const float* x  = (const float*)d_in[0];
  const int*   ei = (const int*)d_in[1];
  const float* ew = (const float*)d_in[2];
  const float* W  = (const float*)d_in[3];
  const float* b  = (const float*)d_in[4];
  float* out = (float*)d_out;

  const int N = in_sizes[0] / D_DIM;
  const int E = in_sizes[2];
  const size_t ND = (size_t)N * D_DIM;

  char* ws = (char*)d_ws;
  float*    xn  = (float*)ws;                      // N*D f32
  float*    agg = xn + ND;                         // N*D f32
  _Float16* Ah  = (_Float16*)(agg + ND);           // N*D f16
  _Float16* Wh  = Ah + ND;                         // D*D f16

  const int TPB = 256;

  // Convert W to f16 once
  {
    size_t n = (size_t)D_DIM * D_DIM;
    cvt_f16_kernel<<<(unsigned)((n + TPB - 1) / TPB), TPB, 0, stream>>>(W, Wh, n);
  }

  const float* cur = x;
  for (int layer = 0; layer < 2; ++layer) {
    // 1) row-sum normalize -> xn
    rownorm_kernel<<<(N + 7) / 8, TPB, 0, stream>>>(cur, xn, N);
    // 2) agg = 0; scatter-add SpMM
    zero4_kernel<<<(unsigned)((ND / 4 + TPB - 1) / TPB), TPB, 0, stream>>>((float4*)agg, ND / 4);
    spmm_kernel<<<(E + 1) / 2, TPB, 0, stream>>>(xn, ei, ew, agg, E);
    // 3) agg -> f16
    cvt_f16_kernel<<<(unsigned)((ND + TPB - 1) / TPB), TPB, 0, stream>>>(agg, Ah, ND);
    // 4) xn = ELU(Ah * Wh^T + b)   [WMMA, 32x64 per wave]
    {
      int waves = (N / 32) * (D_DIM / 64);
      wmma_gemm_abT_kernel<1><<<waves / 8, TPB, 0, stream>>>(Ah, Wh, b, xn, N, D_DIM);
    }
    cur = xn;
  }

  // 5) L2 normalize -> f16 embeddings
  l2norm_f16_kernel<<<(N + 7) / 8, TPB, 0, stream>>>(xn, Ah, N);

  // 6) out = ReLU(Ah * Ah^T)   [WMMA, 8192x8192x512]
  {
    int waves = (N / 32) * (N / 64);
    wmma_gemm_abT_kernel<2><<<waves / 8, TPB, 0, stream>>>(Ah, Ah, b, out, N, N);
  }
}
